// ICLearner_47485158425004
// MI455X (gfx1250) — matrix operand order repaired and checked by reading, not compiled, not run on hardware
//
#include <hip/hip_runtime.h>
#include <math.h>

// ---------------------------------------------------------------------------
// Types / WMMA helpers (CDNA5 / gfx1250, wave32)
// ---------------------------------------------------------------------------
typedef __attribute__((ext_vector_type(16))) _Float16 v16h;
typedef __attribute__((ext_vector_type(8)))  float    v8f;
typedef __attribute__((ext_vector_type(4)))  unsigned int u32x4;
typedef __attribute__((ext_vector_type(8)))  int      i32x8;
typedef __attribute__((ext_vector_type(4)))  int      i32x4;

__device__ __forceinline__ v16h frag2(const _Float16* p0, const _Float16* p1) {
    union { uint4 q[2]; v16h v; } u;
    u.q[0] = *(const uint4*)p0;
    u.q[1] = *(const uint4*)p1;
    return u.v;
}

// A-matrix fragment (16x32 f16): lane-half 0 holds K {0..7,16..23}, half 1 {8..15,24..31}
__device__ __forceinline__ v16h fragA(const _Float16* row, int half) {
    const int k0 = half * 8;
    return frag2(row + k0, row + k0 + 16);
}
// B-matrix fragment (32x16 f16), stored [n][k]: lane-half 0 holds K {0..7,16..23}, half 1 {8..15,24..31}
__device__ __forceinline__ v16h fragB(const _Float16* row, int half) {
    const int k0 = half * 8;
    return frag2(row + k0, row + k0 + 16);
}

__device__ __forceinline__ v8f wmma_f16(v16h a, v16h b, v8f c) {
    return __builtin_amdgcn_wmma_f32_16x16x32_f16(false, a, false, b, (short)0, c, false, false);
}

// ---------------------------------------------------------------------------
// DPP16 in-VALU reductions (no LDS round trips).
// XOR masks {1,2,7,15}: quad_perm(1,0,3,2)=0xB1, quad_perm(2,3,0,1)=0x4E,
// row_half_mirror=0x141 (xor7), row_mirror=0x140 (xor15) span a 16-lane group.
// ---------------------------------------------------------------------------
template <int CTRL>
__device__ __forceinline__ float dpp_f32(float x) {
    union { float f; int i; } a, b;
    a.f = x;
    b.i = __builtin_amdgcn_update_dpp(a.i, a.i, CTRL, 0xf, 0xf, true);
    return b.f;
}
__device__ __forceinline__ float half_max(float x) {      // max over 16-lane half
    x = fmaxf(x, dpp_f32<0xB1>(x));
    x = fmaxf(x, dpp_f32<0x4E>(x));
    x = fmaxf(x, dpp_f32<0x141>(x));
    x = fmaxf(x, dpp_f32<0x140>(x));
    return x;
}
__device__ __forceinline__ float wave_sum(float x) {      // sum over 32 lanes
    x += dpp_f32<0xB1>(x);
    x += dpp_f32<0x4E>(x);
    x += dpp_f32<0x141>(x);
    x += dpp_f32<0x140>(x);
    x += __shfl_xor(x, 16, 32);                           // cross-half step
    return x;
}

// ---------------------------------------------------------------------------
// Tensor Data Mover: 2D tile (global -> LDS) per CDNA5 ISA ch.8 D# layout.
//   tile0   : elements along contiguous dim (data_size = 2 bytes)
//   tile1   : number of rows
//   stride0 : global elements between rows
//   LDS pad : after 2^(pi+1) dwords insert (pa+1) dwords (padded row stride)
// 6-arg builtin form: (u32x4 g0, i32x8 g1, i32x4 g2, i32x4 g3, i32x8, i32 cpol)
// ---------------------------------------------------------------------------
__device__ __forceinline__ void tdm_load_2d(unsigned lds_byte_addr,
                                            const _Float16* gptr,
                                            unsigned tile0, unsigned tile1,
                                            unsigned stride0,
                                            unsigned pi, unsigned pa) {
    const unsigned long long ga = (unsigned long long)(size_t)gptr;
    u32x4 g0 = { 1u,                                  // count=1 (valid descriptor)
                 lds_byte_addr,                       // lds_addr [63:32]
                 (unsigned)ga,                        // global_addr lo
                 (unsigned)((ga >> 32) & 0x01FFFFFFu) | (2u << 30) };  // addr hi | type=2
    const unsigned w0 = (1u << 16)      // data_size = 1 -> 2 bytes
                      | (1u << 20)      // pad_enable
                      | (pi << 22)      // pad_interval
                      | (pa << 25);     // pad_amount
    i32x8 g1 = { (int)w0,
                 (int)((tile0 & 0xFFFFu) << 16),                                  // tensor_dim0 lo
                 (int)(((tile0 >> 16) & 0xFFFFu) | ((tile1 & 0xFFFFu) << 16)),    // dim0 hi | dim1 lo
                 (int)(((tile1 >> 16) & 0xFFFFu) | ((tile0 & 0xFFFFu) << 16)),    // dim1 hi | tile_dim0
                 (int)(tile1 & 0xFFFFu),                                          // tile_dim1 (tile_dim2=0)
                 (int)stride0,                                                    // dim0_stride lo
                 0, 0 };                                                          // stride hi, dim1_stride
    const i32x4 z4 = { 0, 0, 0, 0 };
    const i32x8 z8 = { 0, 0, 0, 0, 0, 0, 0, 0 };
    __builtin_amdgcn_tensor_load_to_lds(g0, g1, z4, z4, z8, 0);
}

// ---------------------------------------------------------------------------
// fp32 -> fp16 conversion (weights)
// ---------------------------------------------------------------------------
__global__ __launch_bounds__(256) void f32_to_f16(const float* __restrict__ s,
                                                  _Float16* __restrict__ d, int n) {
    int i = blockIdx.x * 256 + threadIdx.x;
    int stride = gridDim.x * 256;
    for (; i < n; i += stride) d[i] = (_Float16)s[i];
}

// ---------------------------------------------------------------------------
// LayerNorm over rows of 256, output fp16.  1 wave per row, 8 rows per block.
// ---------------------------------------------------------------------------
__global__ __launch_bounds__(256) void ln_rows(const float* __restrict__ x,
                                               const float* __restrict__ w,
                                               const float* __restrict__ b,
                                               _Float16* __restrict__ out) {
    const int row  = blockIdx.x * 8 + (threadIdx.x >> 5);
    const int lane = threadIdx.x & 31;
    const float* xr = x + (size_t)row * 256 + lane * 8;
    float v[8];
    *(float4*)&v[0] = *(const float4*)(xr);
    *(float4*)&v[4] = *(const float4*)(xr + 4);
    float s = 0.f;
#pragma unroll
    for (int i = 0; i < 8; ++i) s += v[i];
    s = wave_sum(s);
    const float mu = s * (1.0f / 256.0f);
    float vs = 0.f;
#pragma unroll
    for (int i = 0; i < 8; ++i) { float d = v[i] - mu; vs += d * d; }
    vs = wave_sum(vs);
    const float rs = rsqrtf(vs * (1.0f / 256.0f) + 1e-5f);
#pragma unroll
    for (int i = 0; i < 8; ++i) {
        int c = lane * 8 + i;
        out[(size_t)row * 256 + c] = (_Float16)((v[i] - mu) * rs * w[c] + b[c]);
    }
}

// ---------------------------------------------------------------------------
// GEMM  C[M,N] = A[M,K] * W[N,K]^T + bias, WMMA f16 -> f32.
// MODE 0: store fp16.  MODE 1: exact-erf GELU, store fp16.
// MODE 2: residual add into fp32 out (x += C).
// Block = 256 threads = 8 waves.  Block tile 128x64, wave tile 16x64.
// Panels double-buffered through the Tensor Data Mover (wave0: A, wave1: B);
// next round's DMA overlaps current round's WMMAs; TDM ops from one wave
// complete in order, so s_wait_tensorcnt(1) means the current buffer is ready.
// Padded LDS rows (72 halfs = 144B) come from the TDM pad feature.
// ---------------------------------------------------------------------------
template <int MODE>
__global__ __launch_bounds__(256) void gemm_wmma(const _Float16* __restrict__ A,
                                                 const _Float16* __restrict__ W,
                                                 const float* __restrict__ bias,
                                                 void* __restrict__ outp,
                                                 int N, int K) {
    __shared__ _Float16 As[2][128][72];  // 64 data halfs + 8 pad (pi=4 -> 32dw, pa=3 -> 4dw)
    __shared__ _Float16 Bs[2][64][72];
    const int t = threadIdx.x;
    const int m0 = blockIdx.y * 128, n0 = blockIdx.x * 64;
    const int w = t >> 5, lane = t & 31, half = lane >> 4, l15 = lane & 15;

    const unsigned asA0 = (unsigned)(size_t)&As[0][0][0];
    const unsigned asA1 = (unsigned)(size_t)&As[1][0][0];
    const unsigned bsA0 = (unsigned)(size_t)&Bs[0][0][0];
    const unsigned bsA1 = (unsigned)(size_t)&Bs[1][0][0];

    v8f c[4];
#pragma unroll
    for (int j = 0; j < 4; ++j) c[j] = (v8f){};

    // prologue: DMA first panels into buffer 0
    if (w == 0)      tdm_load_2d(asA0, A + (size_t)m0 * K, 64u, 128u, (unsigned)K, 4u, 3u);
    else if (w == 1) tdm_load_2d(bsA0, W + (size_t)n0 * K, 64u, 64u, (unsigned)K, 4u, 3u);

    int cur = 0;
    for (int kk = 0; kk < K; kk += 64) {
        const bool more = (kk + 64) < K;
        if (more) {   // overlap next DMA with this round's compute
            if (w == 0)
                tdm_load_2d(cur ? asA0 : asA1, A + (size_t)m0 * K + kk + 64,
                            64u, 128u, (unsigned)K, 4u, 3u);
            else if (w == 1)
                tdm_load_2d(cur ? bsA0 : bsA1, W + (size_t)n0 * K + kk + 64,
                            64u, 64u, (unsigned)K, 4u, 3u);
        }
        if (w < 2) {
            if (more) __builtin_amdgcn_s_wait_tensorcnt(1);
            else      __builtin_amdgcn_s_wait_tensorcnt(0);
        }
        __syncthreads();
#pragma unroll
        for (int ks = 0; ks < 64; ks += 32) {
            v16h a = fragA(&As[cur][w * 16 + l15][ks], half);
#pragma unroll
            for (int j = 0; j < 4; ++j)
                c[j] = wmma_f16(a, fragB(&Bs[cur][j * 16 + l15][ks], half), c[j]);
        }
        __syncthreads();
        cur ^= 1;
    }

#pragma unroll
    for (int j = 0; j < 4; ++j) {
        const int gn = n0 + j * 16 + l15;
        const float bv = bias[gn];
#pragma unroll
        for (int r = 0; r < 8; ++r) {
            const int gm = m0 + w * 16 + r + half * 8;
            float acc = c[j][r] + bv;
            if (MODE == 1)
                acc = 0.5f * acc * (1.0f + erff(acc * 0.70710678118654752440f));
            if (MODE <= 1) {
                ((_Float16*)outp)[(size_t)gm * N + gn] = (_Float16)acc;
            } else {
                float* xo = (float*)outp;
                const size_t idx = (size_t)gm * N + gn;
                xo[idx] = xo[idx] + acc;
            }
        }
    }
}

// ---------------------------------------------------------------------------
// Flash attention (context-restricted): every query attends to keys [0,n_ctx).
// qkv fp16 [B*S, 768] (q|k|v, H=8 heads of 32). grid (S/64, H, B), 128 thr = 4 waves.
// K panel staged by TDM, V transposed manually, online softmax with per-half
// running max via DPP16 butterflies (no LDS), row-sums via chained P@ones
// WMMAs; hardware __expf; 1/sqrt(hd) folded into the Q fragment.
// ---------------------------------------------------------------------------
__global__ __launch_bounds__(128) void flash_attn(const _Float16* __restrict__ qkv,
                                                  _Float16* __restrict__ o16,
                                                  const int* __restrict__ nctxp) {
    __shared__ _Float16 Ks[64][48];      // 32 data + 16 pad (TDM: pi=3 -> 16dw, pa=7 -> 8dw)
    __shared__ _Float16 Vt[32][72];      // [d][key]   (B layout for P*V)
    __shared__ _Float16 Ps[4][16][72];   // per-wave P tile [qrow][key]

    const int t = threadIdx.x, w = t >> 5, lane = t & 31, half = lane >> 4, l15 = lane & 15;
    const int h = blockIdx.y;
    const size_t tok0 = (size_t)blockIdx.z * 4096;
    const size_t qbase = tok0 + blockIdx.x * 64 + w * 16;
    const int nctx = nctxp[0];
    const unsigned ksAddr = (unsigned)(size_t)&Ks[0][0];

    // Q fragment, pre-scaled by 1/sqrt(hd).
    const _Float16* qp = qkv + (qbase + l15) * 768 + h * 32;
    v16h qa = fragA(qp, half);
    const _Float16 sh = (_Float16)0.17677669529663688f;
#pragma unroll
    for (int i = 0; i < 16; ++i) qa[i] = qa[i] * sh;

    // all-ones B fragment for row-sums
    union { _Float16 hh[16]; v16h v; } uo;
#pragma unroll
    for (int i = 0; i < 16; ++i) uo.hh[i] = (_Float16)1.0f;
    const v16h ones = uo.v;

    float mrun = -3.0e38f;
    v8f lacc = (v8f){};
    v8f oa[2];
    oa[0] = (v8f){}; oa[1] = (v8f){};
    const v8f zero = (v8f){};

    for (int kb = 0; kb < nctx; kb += 64) {
        // K panel via TDM: 64 rows x 32 halfs, global row stride 768 halfs
        if (w == 0)
            tdm_load_2d(ksAddr, qkv + (tok0 + kb) * 768 + 256 + h * 32,
                        32u, 64u, 768u, 3u, 7u);
        // V panel transposed manually
#pragma unroll
        for (int it = 0; it < 2; ++it) {
            const int idx = t + it * 128;
            const int kr = idx >> 2, ch = (idx & 3) * 8;
            union { uint4 q; _Float16 hh[8]; } u;
            u.q = *(const uint4*)(qkv + (tok0 + kb + kr) * 768 + 512 + h * 32 + ch);
#pragma unroll
            for (int i = 0; i < 8; ++i) Vt[ch + i][kr] = u.hh[i];
        }
        if (w == 0) __builtin_amdgcn_s_wait_tensorcnt(0);
        __syncthreads();

        // scores: 16 queries x 64 keys (4 WMMAs), scale already in Q
        v8f s[4];
#pragma unroll
        for (int j = 0; j < 4; ++j)
            s[j] = wmma_f16(qa, fragB(&Ks[j * 16 + l15][0], half), zero);

        // per-half running max via DPP (rows live within one 16-lane half)
        float mx = -3.0e38f;
#pragma unroll
        for (int j = 0; j < 4; ++j)
#pragma unroll
            for (int r = 0; r < 8; ++r) mx = fmaxf(mx, s[j][r]);
        mx = half_max(mx);
        const float mn = fmaxf(mrun, mx);
        const float alpha = __expf(mrun - mn);
        mrun = mn;
#pragma unroll
        for (int r = 0; r < 8; ++r) {
            lacc[r]  *= alpha;
            oa[0][r] *= alpha;
            oa[1][r] *= alpha;
        }

        // probabilities -> LDS (C layout -> A layout redistribution)
#pragma unroll
        for (int j = 0; j < 4; ++j)
#pragma unroll
            for (int r = 0; r < 8; ++r)
                Ps[w][r + 8 * half][j * 16 + l15] = (_Float16)__expf(s[j][r] - mn);
        __syncthreads();

        // O += P*V and row-sums l += P@ones  (2 K-chunks)
        v8f ls = zero;
#pragma unroll
        for (int cc = 0; cc < 2; ++cc) {
            v16h a = fragA(&Ps[w][l15][cc * 32], half);
            ls = wmma_f16(a, ones, ls);
#pragma unroll
            for (int j2 = 0; j2 < 2; ++j2)
                oa[j2] = wmma_f16(a, fragB(&Vt[j2 * 16 + l15][cc * 32], half), oa[j2]);
        }
#pragma unroll
        for (int r = 0; r < 8; ++r) lacc[r] += ls[r];
        __syncthreads();
    }

#pragma unroll
    for (int r = 0; r < 8; ++r) {
        const float inv = 1.0f / lacc[r];
        const size_t tk = qbase + r + 8 * half;
        o16[tk * 256 + h * 32 + l15]      = (_Float16)(oa[0][r] * inv);
        o16[tk * 256 + h * 32 + 16 + l15] = (_Float16)(oa[1][r] * inv);
    }
}

// ---------------------------------------------------------------------------
// Host: full 4-layer forward.  x lives in d_out (fp32, updated in place).
// ---------------------------------------------------------------------------
extern "C" void kernel_launch(void* const* d_in, const int* in_sizes, int n_in,
                              void* d_out, int out_size, void* d_ws, size_t ws_size,
                              hipStream_t stream) {
    (void)in_sizes; (void)n_in; (void)out_size; (void)ws_size;
    const int L = 4, D = 256, FF = 512, NB = 2, S = 4096, TD = 3 * D;
    const int M = NB * S;  // 8192

    const float* seq  = (const float*)d_in[0];
    const float* bqkv = (const float*)d_in[2];
    const float* bo   = (const float*)d_in[4];
    const float* ln1w = (const float*)d_in[5];
    const float* ln1b = (const float*)d_in[6];
    const float* ln2w = (const float*)d_in[7];
    const float* ln2b = (const float*)d_in[8];
    const float* b1   = (const float*)d_in[10];
    const float* b2   = (const float*)d_in[12];
    const int*   nctx = (const int*)d_in[13];

    // carve fp16 workspace (256B-aligned chunks)
    char* wsb = (char*)d_ws;
    size_t off = 0;
    auto carve = [&](size_t elems) -> _Float16* {
        _Float16* p = (_Float16*)(wsb + off);
        off += ((elems * 2 + 255) / 256) * 256;
        return p;
    };
    _Float16* Wqkv16 = carve((size_t)L * TD * D);
    _Float16* Wo16   = carve((size_t)L * D * D);
    _Float16* W116   = carve((size_t)L * FF * D);
    _Float16* W216   = carve((size_t)L * D * FF);
    _Float16* a16    = carve((size_t)M * D);
    _Float16* qkv16  = carve((size_t)M * TD);
    _Float16* o16    = carve((size_t)M * D);
    _Float16* t16    = carve((size_t)M * FF);

    float* x = (float*)d_out;
    (void)hipMemcpyAsync(x, seq, (size_t)M * D * sizeof(float),
                         hipMemcpyDeviceToDevice, stream);

    auto cv = [&](const void* s, _Float16* dptr, size_t n) {
        unsigned g = (unsigned)((n + 2047) / 2048);
        f32_to_f16<<<g, 256, 0, stream>>>((const float*)s, dptr, (int)n);
    };
    cv(d_in[1],  Wqkv16, (size_t)L * TD * D);
    cv(d_in[3],  Wo16,   (size_t)L * D * D);
    cv(d_in[9],  W116,   (size_t)L * FF * D);
    cv(d_in[11], W216,   (size_t)L * D * FF);

    for (int l = 0; l < L; ++l) {
        // h = LN1(x)
        ln_rows<<<M / 8, 256, 0, stream>>>(x, ln1w + l * D, ln1b + l * D, a16);
        // qkv = h @ Wqkv^T + bqkv   (fp16 out)
        gemm_wmma<0><<<dim3(TD / 64, M / 128), 256, 0, stream>>>(
            a16, Wqkv16 + (size_t)l * TD * D, bqkv + l * TD, qkv16, TD, D);
        // o = attention(q,k[:nctx],v[:nctx])
        flash_attn<<<dim3(S / 64, 8, NB), 128, 0, stream>>>(qkv16, o16, nctx);
        // x += o @ Wo^T + bo
        gemm_wmma<2><<<dim3(D / 64, M / 128), 256, 0, stream>>>(
            o16, Wo16 + (size_t)l * D * D, bo + l * D, x, D, D);
        // h2 = LN2(x)
        ln_rows<<<M / 8, 256, 0, stream>>>(x, ln2w + l * D, ln2b + l * D, a16);
        // t = gelu(h2 @ W1^T + b1)  (fp16 out)
        gemm_wmma<1><<<dim3(FF / 64, M / 128), 256, 0, stream>>>(
            a16, W116 + (size_t)l * FF * D, b1 + l * FF, t16, FF, D);
        // x += t @ W2^T + b2
        gemm_wmma<2><<<dim3(D / 64, M / 128), 256, 0, stream>>>(
            t16, W216 + (size_t)l * D * FF, b2 + l * D, x, D, FF);
    }
}
